// Decoder_14791867367535
// MI455X (gfx1250) — matrix operand order repaired and checked by reading, not compiled, not run on hardware
//
#include <hip/hip_runtime.h>
#include <hip/hip_bf16.h>
#include <math.h>

// ---------------------------------------------------------------------------
// Problem constants (from reference)
// ---------------------------------------------------------------------------
#define LQ 48
#define BB 32
#define QQ 256
#define LA 100
#define II 256
#define HH 512
#define AA 256

#define KTOT 1536          // [ctx(256), x_t(256), h_parent(512), h(512)]
#define NGATE 2048         // 4*H
#define NT_G 128           // gate GEMM: N tiles of 16
#define KS_G 48            // gate GEMM: K slabs of 32 (bf16)
#define ASTR 1544          // actIn LDS row stride in bf16 (1536 + 8 pad = +16B)
#define VSTR 260           // V LDS row stride in f32 (256 + 4 pad)

typedef __attribute__((ext_vector_type(16))) __bf16 v16bf;
typedef __attribute__((ext_vector_type(8)))  __bf16 v8bf;
typedef __attribute__((ext_vector_type(8)))  float  v8f;
typedef __attribute__((ext_vector_type(4)))  int    v4i;

// workspace layout (bytes) — total ~12.7 MB of d_ws
#define WB_SZ    ((size_t)NT_G * KS_G * 32 * 16 * 2)     // 6,291,456 packed gate weights
#define W1H_OFF  (WB_SZ)
#define W1H_SZ   ((size_t)16 * 16 * 32 * 16 * 2)         //   262,144 packed W1[:,256:]
#define W1Q_OFF  (W1H_OFF + W1H_SZ)
#define W1Q_SZ   ((size_t)16 * 8 * 32 * 16 * 2)          //   131,072 packed W1[:,:256]
#define BIAS_OFF (W1Q_OFF + W1Q_SZ)
#define BIAS_SZ  ((size_t)NGATE * 4)                     //     8,192 b_ih+b_hh
#define U_OFF    (BIAS_OFF + BIAS_SZ)
#define U_SZ     ((size_t)LQ * BB * AA * 4)              // 1,572,864 U = nlq@W1q.T + b1
#define XB_OFF   (U_OFF + U_SZ)
#define XB_SZ    ((size_t)LA * BB * II * 2)              // 1,638,400 bf16 action feats
#define HB_OFF   (XB_OFF + XB_SZ)
#define HB_SZ    ((size_t)(LA + 1) * BB * HH * 2)        // 3,309,568 bf16 history shadow

// ---------------------------------------------------------------------------
// CDNA5 feature gates (device pass only)
// ---------------------------------------------------------------------------
#if defined(__gfx1250__)
#if __has_builtin(__builtin_amdgcn_global_load_async_to_lds_b128) && \
    __has_builtin(__builtin_amdgcn_s_wait_asynccnt)
#define USE_ASYNC_LDS 1
#endif
#if __has_builtin(__builtin_amdgcn_tanhf)
#define USE_NATIVE_TANH 1
#endif
#endif

__device__ __forceinline__ float fast_tanh(float x) {
#ifdef USE_NATIVE_TANH
  return __builtin_amdgcn_tanhf(x);          // v_tanh_f32 (TRANS, co-executes w/ WMMA)
#else
  return tanhf(x);
#endif
}
__device__ __forceinline__ float sigm(float x) { return 1.0f / (1.0f + __expf(-x)); }

#ifdef USE_ASYNC_LDS
typedef __attribute__((address_space(1))) v4i as1_v4i;   // global
typedef __attribute__((address_space(3))) v4i as3_v4i;   // LDS
__device__ __forceinline__ void async_cp16(const __bf16* g, __bf16* l) {
  void* gg = (void*)g;                       // drop const, stay generic
  void* ll = (void*)l;
  __builtin_amdgcn_global_load_async_to_lds_b128(
      (as1_v4i*)gg, (as3_v4i*)ll, /*offset=*/0, /*cpol=*/0);
}
__device__ __forceinline__ void async_wait() { __builtin_amdgcn_s_wait_asynccnt(0); }
#else
__device__ __forceinline__ void async_cp16(const __bf16* g, __bf16* l) {
  *(v8bf*)l = *(const v8bf*)g;               // fallback: sync 16B copy through VGPRs
}
__device__ __forceinline__ void async_wait() {}
#endif

__device__ __forceinline__ v16bf cat8(v8bf lo, v8bf hi) {
  return __builtin_shufflevector(lo, hi, 0,1,2,3,4,5,6,7,8,9,10,11,12,13,14,15);
}
__device__ __forceinline__ v8f wmma_bf16(v16bf a, v16bf b, v8f c) {
  return __builtin_amdgcn_wmma_f32_16x16x32_bf16(false, a, false, b, (short)0, c, false, false);
}

// ---------------------------------------------------------------------------
// Pack [W_ih | W_hh] (2048 x 1536, fp32) into bf16 WMMA-B fragments:
//   wB[((nt*48 + slab)*32 + lane)*16 + e] = Wcomb[nt*16 + (lane&15)][slab*32 + (lane>>4)*8 + e + (e>=8)*8]
// (B layout: lane holds column N=lane&15; element e <-> K = half*8 + e for e<8, 16 + half*8 + (e-8) for e>=8)
// ---------------------------------------------------------------------------
__global__ void pack_gates_kernel(const float* __restrict__ Wih,
                                  const float* __restrict__ Whh,
                                  __bf16* __restrict__ wB) {
  int idx = blockIdx.x * blockDim.x + threadIdx.x;      // 128*48*32 = 196608 exact
  int lane = idx & 31;
  int slab = (idx >> 5) % KS_G;
  int nt   = idx / (KS_G * 32);
  int half = lane >> 4, nn = lane & 15;
  int n = nt * 16 + nn;
  __bf16* dst = wB + (size_t)idx * 16;
#pragma unroll
  for (int e = 0; e < 16; ++e) {
    int k = slab * 32 + half * 8 + (e < 8 ? e : e + 8);
    float v = (k < 1024) ? Wih[(size_t)n * 1024 + k] : Whh[(size_t)n * 512 + (k - 1024)];
    dst[e] = (__bf16)v;
  }
}

// Pack W1 halves + bias sum (one launch, index ranges).
__global__ void pack_aux_kernel(const float* __restrict__ W1,
                                const float* __restrict__ bih,
                                const float* __restrict__ bhh,
                                __bf16* __restrict__ wB1h,
                                __bf16* __restrict__ wB1q,
                                float* __restrict__ bias) {
  int idx = blockIdx.x * blockDim.x + threadIdx.x;      // 8192 + 4096 + 2048 = 14336 exact
  if (idx < 8192) {                                     // W1[:, 256:768] -> 16 nt x 16 slab
    int lane = idx & 31, slab = (idx >> 5) & 15, nt = idx >> 9;
    int half = lane >> 4, nn = lane & 15;
    __bf16* dst = wB1h + (size_t)idx * 16;
#pragma unroll
    for (int e = 0; e < 16; ++e) {
      int k = slab * 32 + half * 8 + (e < 8 ? e : e + 8); // k in [0,512)
      dst[e] = (__bf16)W1[(size_t)(nt * 16 + nn) * 768 + 256 + k];
    }
  } else if (idx < 8192 + 4096) {                       // W1[:, 0:256] -> 16 nt x 8 slab
    int j = idx - 8192;
    int lane = j & 31, slab = (j >> 5) & 7, nt = j >> 8;
    int half = lane >> 4, nn = lane & 15;
    __bf16* dst = wB1q + (size_t)j * 16;
#pragma unroll
    for (int e = 0; e < 16; ++e) {
      int k = slab * 32 + half * 8 + (e < 8 ? e : e + 8); // k in [0,256)
      dst[e] = (__bf16)W1[(size_t)(nt * 16 + nn) * 768 + k];
    }
  } else {                                              // bias sum (2048)
    int i = idx - 12288;
    bias[i] = bih[i] + bhh[i];
  }
}

// bf16 shadows: action_features -> xf_bf, history[0] -> hist_bf[0]
__global__ void cvt_bf16_kernel(const float* __restrict__ xfeat,
                                const float* __restrict__ hist_in,
                                __bf16* __restrict__ xf_bf,
                                __bf16* __restrict__ hist_bf) {
  int idx = blockIdx.x * blockDim.x + threadIdx.x;      // 819200 + 16384 = 835584 exact
  if (idx < LA * BB * II) {
    xf_bf[idx] = (__bf16)xfeat[idx];
  } else {
    int j = idx - LA * BB * II;                         // j < 16384
    hist_bf[j] = (__bf16)hist_in[j];
  }
}

// ---------------------------------------------------------------------------
// One-time: U[l*32+b][a] = b1[a] + sum_q nlq[l,b,q] * W1[a][q]   (1536 x 256, K=256)
// One wave per 16x16 tile; 96 m-tiles x 16 n-tiles = 1536 waves = 192 blocks x 8 waves.
// ---------------------------------------------------------------------------
__global__ void u_kernel(const float* __restrict__ nlq, const float* __restrict__ b1,
                         const __bf16* __restrict__ wB1q, float* __restrict__ U) {
  int wv = threadIdx.x >> 5, lane = threadIdx.x & 31;
  int half = lane >> 4, nn = lane & 15;
  int tile = blockIdx.x * 8 + wv;                       // 0..1535
  int mt = tile >> 4;                                   // 0..95
  int nt = tile & 15;                                   // 0..15
  int row = mt * 16 + nn;                               // flattened (l,b)

  v8f acc = {0.f,0.f,0.f,0.f,0.f,0.f,0.f,0.f};
  for (int slab = 0; slab < 8; ++slab) {
    // A fragment (16x32 bf16): e<8 -> K = half*8+e ; e>=8 -> K = 16 + half*8 + (e-8)
    const float* p = nlq + (size_t)row * QQ + slab * 32 + half * 8;
    v16bf afr;
#pragma unroll
    for (int e = 0; e < 8; ++e) { afr[e] = (__bf16)p[e]; afr[e + 8] = (__bf16)p[e + 16]; }
    const v16bf bfr = *(const v16bf*)&wB1q[(((size_t)nt * 8 + slab) * 32 + lane) * 16];
    acc = wmma_bf16(afr, bfr, acc);
  }
#pragma unroll
  for (int r = 0; r < 8; ++r) {
    int ro = mt * 16 + half * 8 + r;                    // D: M = half*8 + r
    int a  = nt * 16 + nn;                              //    N = lane&15
    U[(size_t)ro * AA + a] = acc[r] + b1[a];
  }
}

// ---------------------------------------------------------------------------
// Persistent single-WGP decoder: 1024 threads = 32 waves, loops t = 0..99.
// ---------------------------------------------------------------------------
__global__ __launch_bounds__(1024, 1)
void decode_kernel(const float* __restrict__ nlq, const unsigned char* __restrict__ nmask,
                   const int* __restrict__ pidx,
                   const float* __restrict__ hist_in, const float* __restrict__ h0in,
                   const float* __restrict__ c0in,
                   const __bf16* __restrict__ wB, const __bf16* __restrict__ wB1h,
                   const float* __restrict__ biasw, const float* __restrict__ Uw,
                   const __bf16* __restrict__ xf_bf, __bf16* __restrict__ hist_bf,
                   const float* __restrict__ W2g, const float* __restrict__ b2g,
                   float* __restrict__ hs, float* __restrict__ cs,
                   float* __restrict__ hist, float* __restrict__ hn,
                   float* __restrict__ cn) {
  __shared__ __bf16 actIn[BB * ASTR];   // 98,816 B: rows b, cols [ctx|x|h_par|h] bf16
  __shared__ float  Vlds[BB * VSTR];    // 33,280 B: h @ W1h.T
  __shared__ float  e_lds[LQ * BB];     //  6,144 B
  __shared__ float  w_lds[LQ * BB];     //  6,144 B
  __shared__ float  W2s[AA];            //  1,024 B

  const int tid  = threadIdx.x;
  const int wv   = tid >> 5;
  const int lane = tid & 31;
  const int half = lane >> 4;
  const int nn   = lane & 15;

  // gate-GEMM tile assignment (fixed per wave/lane for the whole sequence)
  const int mt  = wv & 1;        // m tile (batch rows 16*mt..)
  const int jt2 = wv >> 1;       // 0..15 -> j tiles {2*jt2, 2*jt2+1}
  const int rowA = mt * 16 + nn; // A row (batch) this lane reads

  // ---- init -------------------------------------------------------------
  if (tid < AA) W2s[tid] = W2g[tid];
  for (int p = tid; p < BB * HH; p += 1024) {           // h(0) -> actIn h-slot (bf16)
    int b = p >> 9, k = p & 511;
    actIn[b * ASTR + 1024 + k] = (__bf16)h0in[p];
    hist[p] = hist_in[p];                               // hist[0] = history[0] (f32 out)
  }
  // c state lives in registers for the entire sequence
  float creg[16];
  float bI[2], bF[2], bG[2], bO[2];
#pragma unroll
  for (int jj = 0; jj < 2; ++jj) {
    int col = (jt2 * 2 + jj) * 16 + nn;
    bI[jj] = biasw[0 * HH + col];
    bF[jj] = biasw[1 * HH + col];
    bG[jj] = biasw[2 * HH + col];
    bO[jj] = biasw[3 * HH + col];
#pragma unroll
    for (int r = 0; r < 8; ++r) {
      int b = mt * 16 + half * 8 + r;
      creg[jj * 8 + r] = c0in[(size_t)b * HH + col];
    }
  }
  const float b2v = b2g[0];
  __threadfence();
  __syncthreads();

  // ---- sequential scan ----------------------------------------------------
  for (int t = 0; t < LA; ++t) {
    // P1: V = h @ W1h.T  (32x256, K=512) — one 16x16 tile per wave
    {
      const int mtv = wv & 1, ntv = wv >> 1;            // 2 x 16 tiles
      const int rv = mtv * 16 + nn;
      v8f vacc = {0.f,0.f,0.f,0.f,0.f,0.f,0.f,0.f};
      for (int slab = 0; slab < 16; ++slab) {
        const __bf16* ab = &actIn[rv * ASTR + 1024 + slab * 32 + half * 8];
        v8bf lo = *(const v8bf*)ab;
        v8bf hi = *(const v8bf*)(ab + 16);
        const v16bf bfr = *(const v16bf*)&wB1h[(((size_t)ntv * 16 + slab) * 32 + lane) * 16];
        vacc = wmma_bf16(cat8(lo, hi), bfr, vacc);
      }
#pragma unroll
      for (int r = 0; r < 8; ++r)
        Vlds[(mtv * 16 + half * 8 + r) * VSTR + ntv * 16 + nn] = vacc[r];
    }
    __syncthreads();

    // P2: e[l,b] = sum_a tanh(U[l,b,a] + V[b,a]) * W2[a] + b2 ; mask
    for (int p = tid; p < LQ * BB; p += 1024) {
      int b = p & 31;
      const float* Up = Uw + (size_t)p * AA;
      const float* Vp = Vlds + b * VSTR;
      float acc = b2v;
      for (int a = 0; a < AA; ++a) acc += fast_tanh(Up[a] + Vp[a]) * W2s[a];
      e_lds[p] = nmask[p] ? acc : -1e9f;
    }
    __syncthreads();

    // P3: softmax over l (one thread per b)
    if (tid < BB) {
      int b = tid;
      float m = -1e30f;
      for (int l = 0; l < LQ; ++l) m = fmaxf(m, e_lds[l * BB + b]);
      float s = 0.f;
      for (int l = 0; l < LQ; ++l) {
        float x = __expf(e_lds[l * BB + b] - m);
        w_lds[l * BB + b] = x; s += x;
      }
      float inv = 1.0f / s;
      for (int l = 0; l < LQ; ++l) w_lds[l * BB + b] *= inv;
    }
    __syncthreads();

    // P4: issue async byte-moves for x_t and h_parent (bf16 shadows), then
    //     overlap the VALU ctx computation with them.
    {
      // x_t slot: 8192 bf16 -> 1024 x 16B async copies (1 per thread)
      int bx = tid >> 5, kx = (tid & 31) * 8;
      async_cp16(xf_bf + (size_t)t * BB * II + tid * 8,
                 &actIn[bx * ASTR + 256 + kx]);
      // h_parent slot: 16384 bf16 -> 2 x 16B async copies per thread (gather)
#pragma unroll
      for (int c = 0; c < 2; ++c) {
        int p8 = tid * 8 + c * 8192;
        int b = p8 >> 9, k = p8 & 511;
        int pi = pidx[t * BB + b];
        async_cp16(hist_bf + ((size_t)pi * BB + b) * HH + k,
                   &actIn[b * ASTR + 512 + k]);
      }
    }
    for (int p = tid; p < BB * QQ; p += 1024) {         // ctx (VALU, overlaps async)
      int b = p >> 8, q = p & 255;
      float s = 0.f;
      for (int l = 0; l < LQ; ++l)
        s += w_lds[l * BB + b] * nlq[((size_t)l * BB + b) * QQ + q];
      cs[((size_t)t * BB + b) * QQ + q] = s;
      actIn[b * ASTR + q] = (__bf16)s;
    }
    async_wait();          // own ASYNCcnt==0; barrier below covers all waves
    __syncthreads();

    // P5: gates = [ctx|x|h_par|h] @ [W_ih|W_hh].T   (32x2048, K=1536)
    v8f acc[8];
#pragma unroll
    for (int q = 0; q < 8; ++q) acc[q] = (v8f){0.f,0.f,0.f,0.f,0.f,0.f,0.f,0.f};
    for (int slab = 0; slab < KS_G; ++slab) {
      const __bf16* ab = &actIn[rowA * ASTR + slab * 32 + half * 8];
      v8bf lo = *(const v8bf*)ab;
      v8bf hi = *(const v8bf*)(ab + 16);
      v16bf afr = cat8(lo, hi);
#pragma unroll
      for (int jj = 0; jj < 2; ++jj)
#pragma unroll
        for (int g = 0; g < 4; ++g) {
          int ntile = g * 32 + jt2 * 2 + jj;            // i,f,g,o blocks 512 cols apart
          const v16bf bfr =
              *(const v16bf*)&wB[(((size_t)ntile * KS_G + slab) * 32 + lane) * 16];
          acc[jj * 4 + g] = wmma_bf16(afr, bfr, acc[jj * 4 + g]);
        }
    }
    __syncthreads();   // all waves done reading actIn h-slot before overwrite

    // LSTM pointwise — each lane owns fixed (b, col) cells; c stays in registers
#pragma unroll
    for (int jj = 0; jj < 2; ++jj) {
      int col = (jt2 * 2 + jj) * 16 + nn;
#pragma unroll
      for (int r = 0; r < 8; ++r) {
        int b = mt * 16 + half * 8 + r;
        float ig = acc[jj * 4 + 0][r] + bI[jj];
        float fg = acc[jj * 4 + 1][r] + bF[jj];
        float gg = acc[jj * 4 + 2][r] + bG[jj];
        float og = acc[jj * 4 + 3][r] + bO[jj];
        float cp = creg[jj * 8 + r];
        float c1 = sigm(fg) * cp + sigm(ig) * fast_tanh(gg);
        float h1 = sigm(og) * fast_tanh(c1);
        creg[jj * 8 + r] = c1;
        size_t bh = (size_t)b * HH + col;
        hs[(size_t)t * BB * HH + bh] = h1;
        hist[(size_t)(t + 1) * BB * HH + bh] = h1;
        hist_bf[(size_t)(t + 1) * BB * HH + bh] = (__bf16)h1;  // shadow for gather
        actIn[b * ASTR + 1024 + col] = (__bf16)h1;             // h(t+1) for next step
        if (t == LA - 1) { hn[bh] = h1; cn[bh] = c1; }
      }
    }
    __threadfence();   // release hist_bf[t+1] for next step's async gather
    __syncthreads();
  }
}

// ---------------------------------------------------------------------------
// Host launcher
// ---------------------------------------------------------------------------
extern "C" void kernel_launch(void* const* d_in, const int* in_sizes, int n_in,
                              void* d_out, int out_size, void* d_ws, size_t ws_size,
                              hipStream_t stream) {
  const float*         nlq     = (const float*)d_in[0];
  const unsigned char* nmask   = (const unsigned char*)d_in[1];  // bool mask
  const float*         xfeat   = (const float*)d_in[2];
  // d_in[3] = act_mask (pass-through, unused in compute)
  const int*           pidx    = (const int*)d_in[4];
  const float*         history = (const float*)d_in[5];
  const float*         h0      = (const float*)d_in[6];
  const float*         c0      = (const float*)d_in[7];
  const float*         Wih     = (const float*)d_in[8];
  const float*         Whh     = (const float*)d_in[9];
  const float*         bih     = (const float*)d_in[10];
  const float*         bhh     = (const float*)d_in[11];
  const float*         W1      = (const float*)d_in[12];
  const float*         b1      = (const float*)d_in[13];
  const float*         W2      = (const float*)d_in[14];
  const float*         b2      = (const float*)d_in[15];

  char* W = (char*)d_ws;                  // needs ~12.7 MB
  __bf16* wB      = (__bf16*)(W);
  __bf16* wB1h    = (__bf16*)(W + W1H_OFF);
  __bf16* wB1q    = (__bf16*)(W + W1Q_OFF);
  float*  bias    = (float*)(W + BIAS_OFF);
  float*  U       = (float*)(W + U_OFF);
  __bf16* xf_bf   = (__bf16*)(W + XB_OFF);
  __bf16* hist_bf = (__bf16*)(W + HB_OFF);

  float* O    = (float*)d_out;            // hs | cs | hist | h_n | c_n
  float* hs   = O;
  float* cs   = hs + (size_t)LA * BB * HH;
  float* hist = cs + (size_t)LA * BB * QQ;
  float* hn   = hist + (size_t)(LA + 1) * BB * HH;
  float* cn   = hn + (size_t)BB * HH;

  pack_gates_kernel<<<768, 256, 0, stream>>>(Wih, Whh, wB);
  pack_aux_kernel<<<56, 256, 0, stream>>>(W1, bih, bhh, wB1h, wB1q, bias);
  cvt_bf16_kernel<<<3264, 256, 0, stream>>>(xfeat, history, xf_bf, hist_bf);
  u_kernel<<<192, 256, 0, stream>>>(nlq, b1, wB1q, U);
  decode_kernel<<<1, 1024, 0, stream>>>(nlq, nmask, pidx, history, h0, c0,
                                        wB, wB1h, bias, U, xf_bf, hist_bf,
                                        W2, b2, hs, cs, hist, hn, cn);
}
